// BatchTopKSAE_49357764165962
// MI455X (gfx1250) — matrix operand order repaired and checked by reading, not compiled, not run on hardware
//
#include <hip/hip_runtime.h>

typedef __attribute__((ext_vector_type(16))) _Float16 v16h;
typedef __attribute__((ext_vector_type(8)))  float    v8f;

#define BATCH   4096
#define DMODEL  1024
#define DSAE    16384
#define NBINS   4096

// ---------------------------------------------------------------------------
// CDNA5 async global->LDS copy (16B per lane) + ASYNCcnt wait
// LDS byte offset = low 32 bits of the generic shared pointer (ISA 10.2).
// ---------------------------------------------------------------------------
__device__ __forceinline__ unsigned to_lds_off(const void* p) {
    return (unsigned)(size_t)p;
}
__device__ __forceinline__ void async_copy16(const _Float16* g, void* lds) {
    unsigned off = to_lds_off(lds);
    asm volatile("global_load_async_to_lds_b128 %0, %1, off"
                 :: "v"(off), "v"(g) : "memory");
}
__device__ __forceinline__ void wait_async() {
    asm volatile("s_wait_asynccnt 0x0" ::: "memory");
}

// ---------------------------------------------------------------------------
// Tiled transpose + fp32 -> f16 convert:  out[C][R] = (f16) in[R][C]
// Block 256 (32x8), 32x32 tiles via LDS.
// ---------------------------------------------------------------------------
__global__ __launch_bounds__(256) void transpose_cvt_kernel(const float* __restrict__ in,
                                                            _Float16* __restrict__ out,
                                                            int R, int C) {
    __shared__ _Float16 tile[32][33];
    const int bx = blockIdx.x * 32;          // C offset
    const int by = blockIdx.y * 32;          // R offset
    const int tx = threadIdx.x & 31;
    const int ty = threadIdx.x >> 5;
#pragma unroll
    for (int i = 0; i < 32; i += 8)
        tile[ty + i][tx] = (_Float16)in[(size_t)(by + ty + i) * C + (bx + tx)];
    __syncthreads();
#pragma unroll
    for (int i = 0; i < 32; i += 8)
        out[(size_t)(bx + ty + i) * R + (by + tx)] = tile[tx][ty + i];
}

// ---------------------------------------------------------------------------
// per-row L2 norm of W_dec (one block per row of 1024)
// ---------------------------------------------------------------------------
__global__ __launch_bounds__(256) void rownorm_kernel(const float* __restrict__ W,
                                                      float* __restrict__ norms) {
    __shared__ float red[256];
    const int row = blockIdx.x;
    const int t = threadIdx.x;
    float4 v = ((const float4*)(W + (size_t)row * DMODEL))[t];
    red[t] = v.x * v.x + v.y * v.y + v.z * v.z + v.w * v.w;
    __syncthreads();
    for (int off = 128; off > 0; off >>= 1) {
        if (t < off) red[t] += red[t + off];
        __syncthreads();
    }
    if (t == 0) norms[row] = sqrtf(red[0]);
}

// ---------------------------------------------------------------------------
// diff = LN(LN(tgt) - LN(src)), output f16.  One block per batch row.
// ---------------------------------------------------------------------------
__global__ __launch_bounds__(256) void layernorm_kernel(const float* __restrict__ x,
                                                        _Float16* __restrict__ diff) {
    __shared__ float4 red[256];
    const int row = blockIdx.x;
    const int t = threadIdx.x;
    const float* src = x + (size_t)row * (2 * DMODEL);
    const float* tgt = src + DMODEL;
    float4 s4 = ((const float4*)src)[t];
    float4 t4 = ((const float4*)tgt)[t];
    red[t] = make_float4(s4.x + s4.y + s4.z + s4.w,
                         s4.x * s4.x + s4.y * s4.y + s4.z * s4.z + s4.w * s4.w,
                         t4.x + t4.y + t4.z + t4.w,
                         t4.x * t4.x + t4.y * t4.y + t4.z * t4.z + t4.w * t4.w);
    __syncthreads();
    for (int off = 128; off > 0; off >>= 1) {
        if (t < off) {
            float4 a = red[t], b = red[t + off];
            red[t] = make_float4(a.x + b.x, a.y + b.y, a.z + b.z, a.w + b.w);
        }
        __syncthreads();
    }
    float4 tot = red[0];
    const float inv = 1.0f / DMODEL;
    float ms = tot.x * inv, vs = tot.y * inv - ms * ms;
    float mt = tot.z * inv, vt = tot.w * inv - mt * mt;
    float rs = rsqrtf(vs + 1e-8f), rt = rsqrtf(vt + 1e-8f);
    float d0 = (t4.x - mt) * rt - (s4.x - ms) * rs;
    float d1 = (t4.y - mt) * rt - (s4.y - ms) * rs;
    float d2 = (t4.z - mt) * rt - (s4.z - ms) * rs;
    float d3 = (t4.w - mt) * rt - (s4.w - ms) * rs;
    __syncthreads();
    red[t] = make_float4(d0 + d1 + d2 + d3,
                         d0 * d0 + d1 * d1 + d2 * d2 + d3 * d3, 0.f, 0.f);
    __syncthreads();
    for (int off = 128; off > 0; off >>= 1) {
        if (t < off) {
            float4 a = red[t], b = red[t + off];
            red[t] = make_float4(a.x + b.x, a.y + b.y, 0.f, 0.f);
        }
        __syncthreads();
    }
    float md = red[0].x * inv;
    float vd = red[0].y * inv - md * md;
    float rd = rsqrtf(vd + 1e-8f);
    _Float16* op = diff + (size_t)row * DMODEL + t * 4;
    op[0] = (_Float16)((d0 - md) * rd);
    op[1] = (_Float16)((d1 - md) * rd);
    op[2] = (_Float16)((d2 - md) * rd);
    op[3] = (_Float16)((d3 - md) * rd);
}

// ---------------------------------------------------------------------------
__global__ void zero_hist_kernel(unsigned int* __restrict__ hist) {
    hist[blockIdx.x * 256 + threadIdx.x] = 0u;
}

// ---------------------------------------------------------------------------
// WMMA GEMM: C[M,N] = A[M,K](f16,row) @ Bt[N,K](f16,row)^T + bias
// Both operands K-contiguous -> all tile loads are async b128 -> LDS,
// double-buffered, one barrier per K-step.
// MODE 0 (encoder): relu, store fp32 acts, fused LDS score histogram.
// MODE 1 (decoder): bias epilogue to fp32 out.
// ---------------------------------------------------------------------------
template <int MODE>
__global__ __launch_bounds__(256) void gemm_wmma_kernel(
    const _Float16* __restrict__ A, const _Float16* __restrict__ Bt,
    const float* __restrict__ bias, float* __restrict__ C,
    const float* __restrict__ dnorms, unsigned int* __restrict__ hist,
    int M, int N, int K) {
    __shared__ __align__(16) _Float16 As[2][128][40];   // 2 x 10 KB
    __shared__ __align__(16) _Float16 Bs[2][128][40];   // rows are N, K-contig
    __shared__ unsigned int shist[NBINS];               // 16 KB

    const int t = threadIdx.x;
    const int w = t >> 5;
    const int l = t & 31;
    const int m0 = blockIdx.y * 128;
    const int n0 = blockIdx.x * 128;

    if (MODE == 0)
        for (int i = t; i < NBINS; i += 256) shist[i] = 0u;

    v8f acc[8];
    const v8f vzero = {0.f, 0.f, 0.f, 0.f, 0.f, 0.f, 0.f, 0.f};
#pragma unroll
    for (int s = 0; s < 8; ++s) acc[s] = vzero;

    // per-thread tile-load slice: 16B chunk (row r, k-octet kk), rows r and r+64
    const int mm = t >> 2;
    const int kk = (t & 3) << 3;
    const _Float16* ga0 = A  + (size_t)(m0 + mm) * K + kk;
    const _Float16* ga1 = ga0 + (size_t)64 * K;
    const _Float16* gb0 = Bt + (size_t)(n0 + mm) * K + kk;
    const _Float16* gb1 = gb0 + (size_t)64 * K;

    auto issue_tile = [&](int b) {
        async_copy16(ga0, &As[b][mm][kk]);
        async_copy16(ga1, &As[b][mm + 64][kk]);
        async_copy16(gb0, &Bs[b][mm][kk]);
        async_copy16(gb1, &Bs[b][mm + 64][kk]);
        ga0 += 32; ga1 += 32; gb0 += 32; gb1 += 32;
    };

    // fragment lane mapping (ISA 7.12.2, 16-bit 16x32 A):
    const int lk = (l & 16) ? 8 : 0;        // A: lanes16-31 -> K[8..15]/K[24..31]
    const int bk = (l & 16) ? 8 : 0;        // B: lanes16-31 -> K[8..15]/K[24..31]
    const int ln15 = l & 15;
    const int am = (w << 4) + ln15;

    const int nk = K >> 5;
    issue_tile(0);
    union Frag { uint4 u[2]; v16h h; };
    for (int kt = 0; kt < nk; ++kt) {
        const int buf = kt & 1;
        wait_async();
        __syncthreads();
        if (kt + 1 < nk) issue_tile(buf ^ 1);

        Frag fa;
        fa.u[0] = *(const uint4*)&As[buf][am][lk];
        fa.u[1] = *(const uint4*)&As[buf][am][16 + lk];
        Frag fb[8];
#pragma unroll
        for (int s = 0; s < 8; ++s) {
            const int bn = (s << 4) + ln15;
            fb[s].u[0] = *(const uint4*)&Bs[buf][bn][bk];
            fb[s].u[1] = *(const uint4*)&Bs[buf][bn][bk + 16];
        }
#pragma unroll
        for (int s = 0; s < 8; ++s)
            acc[s] = __builtin_amdgcn_wmma_f32_16x16x32_f16(
                false, fa.h, false, fb[s].h, (short)0, acc[s], false, false);
    }

    // epilogue (C layout: lanes 0-15 N=lane,M=v; lanes 16-31 N=lane-16,M=v+8)
    const int mbase = m0 + (w << 4) + ((l & 16) ? 8 : 0);
#pragma unroll
    for (int s = 0; s < 8; ++s) {
        const int n = n0 + (s << 4) + ln15;
        const float bn = bias[n];
        const float dn = (MODE == 0) ? dnorms[n] : 0.f;
#pragma unroll
        for (int v = 0; v < 8; ++v) {
            const int m = mbase + v;
            float val = acc[s][v] + bn;
            if (MODE == 0) {
                val = fmaxf(val, 0.f);
                C[(size_t)m * N + n] = val;
                const float sc = val * dn;
                const unsigned bin = (sc > 0.f) ? (__float_as_uint(sc) >> 19) : 0u;
                atomicAdd(&shist[bin], 1u);   // branchless; bin 0 ignored later
            } else {
                C[(size_t)m * N + n] = val;
            }
        }
    }
    if (MODE == 0) {
        __syncthreads();
        for (int i = t; i < NBINS; i += 256) {
            const unsigned int c = shist[i];
            if (c) atomicAdd(&hist[i], c);
        }
    }
}

// ---------------------------------------------------------------------------
// Threshold: largest bin b>=1 whose suffix count reaches k*BATCH (else 1)
// ---------------------------------------------------------------------------
__global__ void find_thr_kernel(const unsigned int* __restrict__ hist,
                                const int* __restrict__ kptr,
                                unsigned int* __restrict__ sel) {
    if (threadIdx.x == 0) {
        const unsigned int need = (unsigned int)(*kptr) * (unsigned int)BATCH;
        unsigned int total = 0, s = 1;
        for (int b = NBINS - 1; b >= 1; --b) {
            total += hist[b];
            if (total >= need) { s = (unsigned int)b; break; }
        }
        *sel = s;
    }
}

// ---------------------------------------------------------------------------
// Apply top-k mask + fp32 -> f16 downcast (vectorized x4)
// ---------------------------------------------------------------------------
__device__ __forceinline__ _Float16 keep_f16(float a, float dn, unsigned s) {
    const float sc = a * dn;
    const bool keep = (sc > 0.f) && ((__float_as_uint(sc) >> 19) >= s);
    return keep ? (_Float16)a : (_Float16)0.f;
}
__global__ __launch_bounds__(256) void mask_cvt_kernel(const float4* __restrict__ acts4,
                                                       const float* __restrict__ dnorms,
                                                       const unsigned int* __restrict__ sel,
                                                       ushort4* __restrict__ out4,
                                                       size_t n4) {
    const unsigned int s = *sel;
    size_t i = (size_t)blockIdx.x * blockDim.x + threadIdx.x;
    const size_t stride = (size_t)gridDim.x * blockDim.x;
    for (; i < n4; i += stride) {
        const float4 a = acts4[i];
        const float* dn = dnorms + (unsigned)((i << 2) & (DSAE - 1));
        union { _Float16 h[4]; ushort4 u; } pk;
        pk.h[0] = keep_f16(a.x, dn[0], s);
        pk.h[1] = keep_f16(a.y, dn[1], s);
        pk.h[2] = keep_f16(a.z, dn[2], s);
        pk.h[3] = keep_f16(a.w, dn[3], s);
        out4[i] = pk.u;
    }
}

// ---------------------------------------------------------------------------
extern "C" void kernel_launch(void* const* d_in, const int* in_sizes, int n_in,
                              void* d_out, int out_size, void* d_ws, size_t ws_size,
                              hipStream_t stream) {
    const float* x     = (const float*)d_in[0];   // [4096, 2048]
    const float* W_enc = (const float*)d_in[1];   // [1024, 16384]
    const float* b_enc = (const float*)d_in[2];   // [16384]
    const float* W_dec = (const float*)d_in[3];   // [16384, 1024]
    const float* b_dec = (const float*)d_in[4];   // [1024]
    const int*   kptr  = (const int*)d_in[5];     // scalar k
    float* out = (float*)d_out;                   // [4096, 1024]

    char* ws = (char*)d_ws;
    size_t off = 0;
    _Float16* wencT  = (_Float16*)(ws + off); off += (size_t)DSAE * DMODEL * 2;   // [16384][1024]
    _Float16* wdecT  = (_Float16*)(ws + off); off += (size_t)DMODEL * DSAE * 2;   // [1024][16384]
    _Float16* diff_h = (_Float16*)(ws + off); off += (size_t)BATCH * DMODEL * 2;
    _Float16* actsm  = (_Float16*)(ws + off); off += (size_t)BATCH * DSAE * 2;
    float*    acts   = (float*)(ws + off);    off += (size_t)BATCH * DSAE * 4;
    float*    dnorms = (float*)(ws + off);    off += (size_t)DSAE * 4;
    unsigned int* hist = (unsigned int*)(ws + off); off += NBINS * 4;
    unsigned int* sel  = (unsigned int*)(ws + off); off += 256;

    // 1) W_enc [1024][16384] -> wencT [16384][1024] f16
    transpose_cvt_kernel<<<dim3(DSAE / 32, DMODEL / 32), 256, 0, stream>>>(
        W_enc, wencT, DMODEL, DSAE);
    // 2) W_dec [16384][1024] -> wdecT [1024][16384] f16
    transpose_cvt_kernel<<<dim3(DMODEL / 32, DSAE / 32), 256, 0, stream>>>(
        W_dec, wdecT, DSAE, DMODEL);
    // 3) W_dec row norms
    rownorm_kernel<<<DSAE, 256, 0, stream>>>(W_dec, dnorms);
    // 4) double layernorm -> diff (f16)
    layernorm_kernel<<<BATCH, 256, 0, stream>>>(x, diff_h);
    // 5) zero histogram
    zero_hist_kernel<<<NBINS / 256, 256, 0, stream>>>(hist);
    // 6) encoder GEMM + relu + fused score histogram
    gemm_wmma_kernel<0><<<dim3(DSAE / 128, BATCH / 128), 256, 0, stream>>>(
        diff_h, wencT, b_enc, acts, dnorms, hist, BATCH, DSAE, DMODEL);
    // 7) threshold bin
    find_thr_kernel<<<1, 32, 0, stream>>>(hist, kptr, sel);
    // 8) mask + downcast
    mask_cvt_kernel<<<8192, 256, 0, stream>>>((const float4*)acts, dnorms, sel,
                                              (ushort4*)actsm,
                                              (size_t)BATCH * DSAE / 4);
    // 9) decoder GEMM -> recon
    gemm_wmma_kernel<1><<<dim3(DMODEL / 128, BATCH / 128), 256, 0, stream>>>(
        actsm, wdecT, b_dec, out, nullptr, nullptr, BATCH, DMODEL, DSAE);
}
